// DocREModel_32280974196898
// MI455X (gfx1250) — matrix operand order repaired and checked by reading, not compiled, not run on hardware
//
#include <hip/hip_runtime.h>
#include <cmath>

#define B_  4
#define S_  512
#define D_  768
#define H_  12
#define NE_ 42
#define M_  8
#define OFF_ 1
#define NN_   (NE_*NE_)        /* 1764 rows of the GEMM A matrix */
#define MSUP_ 28               /* supertiles of 64 rows           */
#define MPAD_ (MSUP_*64)       /* 1792 padded rows                */
#define NT_   (D_/16)          /* 48 N-tiles                      */

typedef __attribute__((ext_vector_type(16))) __bf16 v16bf;
typedef __attribute__((ext_vector_type(8)))  float  v8f;

union Frag { v16bf v; uint4 u[2]; };

// ---------------- Kernel 1: ent_emb = masked logsumexp over mentions -------
__global__ void k_ent_emb(const float* __restrict__ seq,
                          const int* __restrict__ pos,
                          const int* __restrict__ mask,
                          float* __restrict__ ent_emb) {
  int be = blockIdx.x;            // b*NE + e
  int b  = be / NE_;
  __shared__ int spos[M_];
  __shared__ int sval[M_];
  if (threadIdx.x < M_) {
    int p  = pos [be*M_ + threadIdx.x] + OFF_;
    int mk = mask[be*M_ + threadIdx.x];
    int v  = (mk > 0) && (p < S_);
    p = p < 0 ? 0 : (p > S_-1 ? S_-1 : p);
    spos[threadIdx.x] = p;
    sval[threadIdx.x] = v;
  }
  __syncthreads();
  const float* sb = seq + (size_t)b * S_ * D_;
  for (int d = threadIdx.x; d < D_; d += blockDim.x) {
    float vals[M_];
    float mx = -INFINITY;
    #pragma unroll
    for (int m = 0; m < M_; ++m) {
      float x = sval[m] ? sb[(size_t)spos[m]*D_ + d] : -INFINITY;
      vals[m] = x;
      mx = fmaxf(mx, x);
    }
    float r;
    if (mx == -INFINITY) {
      r = 0.0f;                       // no valid mention -> 0
    } else {
      float s = 0.0f;
      #pragma unroll
      for (int m = 0; m < M_; ++m)
        if (sval[m]) s += __expf(vals[m] - mx);
      r = mx + __logf(s);
    }
    ent_emb[(size_t)be*D_ + d] = r;
  }
}

// ---------------- Kernel 2: ent_att = masked mean of attention rows --------
__global__ void k_ent_att(const float* __restrict__ att,
                          const int* __restrict__ pos,
                          const int* __restrict__ mask,
                          float* __restrict__ ent_att) {
  int idx = blockIdx.x;             // (b*NE + e)*H + h
  int h  = idx % H_;
  int be = idx / H_;
  int b  = be / NE_;
  int spos[M_], sval[M_], cnt = 0;
  #pragma unroll
  for (int m = 0; m < M_; ++m) {
    int p  = pos [be*M_ + m] + OFF_;
    int mk = mask[be*M_ + m];
    int v  = (mk > 0) && (p < S_);
    p = p < 0 ? 0 : (p > S_-1 ? S_-1 : p);
    spos[m] = p; sval[m] = v; cnt += v;
  }
  float inv = 1.0f / (float)(cnt > 0 ? cnt : 1);
  int s = threadIdx.x;              // blockDim = S_
  const float* ab = att + ((size_t)b * H_ + h) * (size_t)S_ * S_;
  float acc = 0.0f;
  #pragma unroll
  for (int m = 0; m < M_; ++m)
    if (sval[m]) acc += ab[(size_t)spos[m]*S_ + s];
  ent_att[((size_t)be*H_ + h)*S_ + s] = acc * inv;
}

// ---------------- Kernel 3: LDS-tiled transpose+convert seq -> seqT (bf16) -
__global__ void k_seqT(const float* __restrict__ seq,
                       __bf16* __restrict__ seqT) {
  __shared__ float tile[32][33];
  int b  = blockIdx.z;
  int s0 = blockIdx.x * 32;
  int d0 = blockIdx.y * 32;
  const float* sb = seq + (size_t)b * S_ * D_;
  #pragma unroll
  for (int r = threadIdx.y; r < 32; r += 8)
    tile[r][threadIdx.x] = sb[(size_t)(s0 + r) * D_ + d0 + threadIdx.x];
  __syncthreads();
  __bf16* ob = seqT + (size_t)b * D_ * S_;
  #pragma unroll
  for (int r = threadIdx.y; r < 32; r += 8)
    ob[(size_t)(d0 + r) * S_ + s0 + threadIdx.x] = (__bf16)tile[threadIdx.x][r];
}

// ---------------- Kernel 4: ht[b,i,j,s] normalized, stored bf16, padded ----
__global__ void k_ht(const float* __restrict__ ent_att,
                     __bf16* __restrict__ htb) {
  int r  = blockIdx.x;              // b*MPAD_ + row
  int b  = r / MPAD_;
  int ij = r % MPAD_;
  if (ij >= NN_) {                  // zero padding rows (uniform per block)
    #pragma unroll
    for (int t = 0; t < 2; ++t)
      htb[((size_t)b*MPAD_ + ij)*S_ + threadIdx.x + t*256] = (__bf16)0.0f;
    return;
  }
  int i  = ij / NE_, j = ij % NE_;
  const float* ea_i = ent_att + ((size_t)b*NE_ + i) * H_ * S_;
  const float* ea_j = ent_att + ((size_t)b*NE_ + j) * H_ * S_;
  float v[2];
  float local = 0.0f;
  #pragma unroll
  for (int t = 0; t < 2; ++t) {
    int s = threadIdx.x + t*256;
    float acc = 0.0f;
    #pragma unroll
    for (int h = 0; h < H_; ++h)
      acc += ea_i[h*S_ + s] * ea_j[h*S_ + s];
    acc *= (1.0f / (float)H_);
    v[t] = acc;
    local += acc;
  }
  __shared__ float red[8];
  #pragma unroll
  for (int off = 16; off > 0; off >>= 1)
    local += __shfl_down(local, off, 32);
  int wave = threadIdx.x >> 5, lane = threadIdx.x & 31;
  if (lane == 0) red[wave] = local;
  __syncthreads();
  if (threadIdx.x == 0) {
    float z = 0.0f;
    #pragma unroll
    for (int w = 0; w < 8; ++w) z += red[w];
    red[0] = z;
  }
  __syncthreads();
  float invZ = 1.0f / (red[0] + 1e-5f);
  #pragma unroll
  for (int t = 0; t < 2; ++t) {
    int s = threadIdx.x + t*256;
    htb[((size_t)b*MPAD_ + ij)*S_ + s] = (__bf16)(v[t] * invZ);
  }
}

// ---------------- Kernel 5: broadcast hs/ts slabs of the output ------------
__global__ void k_hsts(const float* __restrict__ ent_emb,
                       float* __restrict__ out) {
  size_t idx = (size_t)blockIdx.x * blockDim.x + threadIdx.x;
  size_t total = (size_t)B_ * NN_ * D_;
  if (idx >= total) return;
  int d  = (int)(idx % D_);
  size_t r = idx / D_;
  int ij = (int)(r % NN_);
  int b  = (int)(r / NN_);
  int i  = ij / NE_, j = ij % NE_;
  float hv = ent_emb[((size_t)b*NE_ + i)*D_ + d];
  float tv = ent_emb[((size_t)b*NE_ + j)*D_ + d];
  float* o = out + ((size_t)b*NN_ + ij) * (size_t)(3*D_);
  o[d]       = hv;   // hs
  o[D_ + d]  = tv;   // ts
}

// ---------------- Kernel 6: rs = ht @ seq via V_WMMA_F32_16X16X32_BF16 -----
// One wave32 computes a 64x16 slab (4 M-tiles, shared B fragment).
// Software-pipelined: fragments for k+32 are loaded before the WMMAs for k.
__device__ __forceinline__ void load_frags(const uint4* __restrict__ A0,
                                           const uint4* __restrict__ A1,
                                           const uint4* __restrict__ A2,
                                           const uint4* __restrict__ A3,
                                           const uint4* __restrict__ Bp,
                                           int k0, int hl,
                                           Frag& a0, Frag& a1, Frag& a2,
                                           Frag& a3, Frag& bfr) {
  int ca = (k0 >> 3) + hl;        // A: lo chunk ca, hi chunk ca+2
  int cb = (k0 >> 3) + hl;        // B: lo chunk cb, hi chunk cb+2
  a0.u[0] = A0[ca]; a0.u[1] = A0[ca+2];
  a1.u[0] = A1[ca]; a1.u[1] = A1[ca+2];
  a2.u[0] = A2[ca]; a2.u[1] = A2[ca+2];
  a3.u[0] = A3[ca]; a3.u[1] = A3[ca+2];
  bfr.u[0] = Bp[cb]; bfr.u[1] = Bp[cb+2];
}

__global__ void k_gemm_wmma(const __bf16* __restrict__ htb,   // (B, MPAD_, S)
                            const __bf16* __restrict__ seqT,  // (B, D, S)
                            float* __restrict__ out) {
  int nt = blockIdx.x;              // 0..NT_-1
  int ms = blockIdx.y;              // 0..MSUP_-1
  int b  = blockIdx.z;
  int lane = threadIdx.x & 31;
  int hl = lane >> 4;               // lane half
  int lr = lane & 15;

  const uint4* A0 = (const uint4*)(htb + ((size_t)b*MPAD_ + ms*64 +  0 + lr)*S_);
  const uint4* A1 = (const uint4*)(htb + ((size_t)b*MPAD_ + ms*64 + 16 + lr)*S_);
  const uint4* A2 = (const uint4*)(htb + ((size_t)b*MPAD_ + ms*64 + 32 + lr)*S_);
  const uint4* A3 = (const uint4*)(htb + ((size_t)b*MPAD_ + ms*64 + 48 + lr)*S_);
  int col = nt*16 + lr;
  const uint4* Bp = (const uint4*)(seqT + ((size_t)b*D_ + col)*S_);

  v8f acc0 = {}, acc1 = {}, acc2 = {}, acc3 = {};

  Frag ca0, ca1, ca2, ca3, cbf;
  load_frags(A0, A1, A2, A3, Bp, 0, hl, ca0, ca1, ca2, ca3, cbf);

  #pragma unroll 3
  for (int k0 = 0; k0 < S_ - 32; k0 += 32) {
    Frag na0, na1, na2, na3, nbf;
    load_frags(A0, A1, A2, A3, Bp, k0 + 32, hl, na0, na1, na2, na3, nbf);
    acc0 = __builtin_amdgcn_wmma_f32_16x16x32_bf16(false, ca0.v, false, cbf.v, (short)0, acc0, false, false);
    acc1 = __builtin_amdgcn_wmma_f32_16x16x32_bf16(false, ca1.v, false, cbf.v, (short)0, acc1, false, false);
    acc2 = __builtin_amdgcn_wmma_f32_16x16x32_bf16(false, ca2.v, false, cbf.v, (short)0, acc2, false, false);
    acc3 = __builtin_amdgcn_wmma_f32_16x16x32_bf16(false, ca3.v, false, cbf.v, (short)0, acc3, false, false);
    ca0 = na0; ca1 = na1; ca2 = na2; ca3 = na3; cbf = nbf;
  }
  // epilogue (last k-step)
  acc0 = __builtin_amdgcn_wmma_f32_16x16x32_bf16(false, ca0.v, false, cbf.v, (short)0, acc0, false, false);
  acc1 = __builtin_amdgcn_wmma_f32_16x16x32_bf16(false, ca1.v, false, cbf.v, (short)0, acc1, false, false);
  acc2 = __builtin_amdgcn_wmma_f32_16x16x32_bf16(false, ca2.v, false, cbf.v, (short)0, acc2, false, false);
  acc3 = __builtin_amdgcn_wmma_f32_16x16x32_bf16(false, ca3.v, false, cbf.v, (short)0, acc3, false, false);

  // D layout: VGPR r -> M = hl*8 + r, N = lr
  #pragma unroll
  for (int t = 0; t < 4; ++t) {
    v8f acc = t == 0 ? acc0 : t == 1 ? acc1 : t == 2 ? acc2 : acc3;
    #pragma unroll
    for (int r = 0; r < 8; ++r) {
      int row = ms*64 + t*16 + hl*8 + r;
      if (row < NN_)
        out[((size_t)b*NN_ + row) * (size_t)(3*D_) + 2*D_ + col] = acc[r];
    }
  }
}

extern "C" void kernel_launch(void* const* d_in, const int* in_sizes, int n_in,
                              void* d_out, int out_size, void* d_ws, size_t ws_size,
                              hipStream_t stream) {
  const float* seq  = (const float*)d_in[0];   // (B,S,D)
  const float* att  = (const float*)d_in[1];   // (B,H,S,S)
  const int*   pos  = (const int*)  d_in[2];   // (B,NE,M)
  const int*   mask = (const int*)  d_in[3];   // (B,NE,M)
  float* out = (float*)d_out;                  // (B,NE,NE,3D)

  float*  ent_emb = (float*)d_ws;                               // B*NE*D f32
  float*  ent_att = ent_emb + (size_t)B_*NE_*D_;                // B*NE*H*S f32
  __bf16* seqT    = (__bf16*)(ent_att + (size_t)B_*NE_*H_*S_);  // B*D*S bf16
  __bf16* htb     = seqT + (size_t)B_*D_*S_;                    // B*MPAD_*S bf16

  k_ent_emb<<<B_*NE_,    256, 0, stream>>>(seq, pos, mask, ent_emb);
  k_ent_att<<<B_*NE_*H_, S_,  0, stream>>>(att, pos, mask, ent_att);

  dim3 tg(S_/32, D_/32, B_);
  k_seqT<<<tg, dim3(32, 8), 0, stream>>>(seq, seqT);

  k_ht<<<B_*MPAD_, 256, 0, stream>>>(ent_att, htb);

  {
    size_t total = (size_t)B_*NN_*D_;
    int threads = 256;
    int blocks = (int)((total + threads - 1) / threads);
    k_hsts<<<blocks, threads, 0, stream>>>(ent_emb, out);
  }

  dim3 grid(NT_, MSUP_, B_);
  k_gemm_wmma<<<grid, 32, 0, stream>>>(htb, seqT, out);
}
